// PointNet2SemSeg_68281390072555
// MI455X (gfx1250) — matrix operand order repaired and checked by reading, not compiled, not run on hardware
//
#include <hip/hip_runtime.h>
#include <hip/hip_bf16.h>

typedef __attribute__((ext_vector_type(16))) _Float16 v16h;
typedef __attribute__((ext_vector_type(8)))  _Float16 v8h;
typedef __attribute__((ext_vector_type(8)))  float    v8f;
typedef __attribute__((ext_vector_type(4)))  float    v4f;

#define BN_INV_STD 0.999995000037499f  // 1/sqrt(1+1e-5), eval-mode BatchNorm

// ---------------------------------------------------------------------------
// WMMA helper (CDNA5 wave32, v_wmma_f32_16x16x32_f16, f32 accumulate)
// ---------------------------------------------------------------------------
__device__ __forceinline__ v8f wmma32(v16h a, v16h b, v8f c) {
  return __builtin_amdgcn_wmma_f32_16x16x32_f16(false, a, false, b, (short)0, c,
                                                false, false);
}

// ---------------------------------------------------------------------------
// Fragment loaders (wave32 WMMA layouts per CDNA5 ISA).
// ---------------------------------------------------------------------------

// A from global f32 row-major, fully in-bounds, lda % 4 == 0 -> 4x b128 loads
__device__ __forceinline__ v16h a_vec(const float* __restrict__ A, int lda,
                                      int row0, int k0) {
  int lane = threadIdx.x & 31;
  const float* p = A + (size_t)(row0 + (lane & 15)) * lda + k0 + ((lane >> 4) << 3);
  v4f f0 = *(const v4f*)(p);
  v4f f1 = *(const v4f*)(p + 4);
  v4f f2 = *(const v4f*)(p + 16);
  v4f f3 = *(const v4f*)(p + 20);
  v16h a;
#pragma unroll
  for (int i = 0; i < 4; ++i) {
    a[i]      = (_Float16)f0[i];
    a[4 + i]  = (_Float16)f1[i];
    a[8 + i]  = (_Float16)f2[i];
    a[12 + i] = (_Float16)f3[i];
  }
  return a;
}

// A from global f32, clamped-address branchless safe path (generic fallback)
__device__ __forceinline__ v16h a_safe(const float* __restrict__ A, int lda,
                                       int row0, int k0, int M, int K) {
  int lane = threadIdx.x & 31;
  int r = row0 + (lane & 15);
  int rc = (r < M) ? r : (M - 1);
  bool rok = (r < M);
  int kg = (lane >> 4) << 3;
  const float* p = A + (size_t)rc * lda;
  v16h a;
#pragma unroll
  for (int q = 0; q < 16; ++q) {
    int kb = k0 + ((q >> 3) << 4) + kg + (q & 7);
    int kc = (kb < K) ? kb : (K - 1);
    float x = p[kc];
    a[q] = (_Float16)((rok && kb < K) ? x : 0.f);
  }
  return a;
}

// B = W^T from f32 activations (K matrix), fully in-bounds, ldw % 4 == 0
__device__ __forceinline__ v16h b_wT_vec(const float* __restrict__ W, int ldw,
                                         int col0, int k0) {
  int lane = threadIdx.x & 31;
  const float* p = W + (size_t)(col0 + (lane & 15)) * ldw + k0 + ((lane >> 4) << 3);
  v4f f0 = *(const v4f*)(p);
  v4f f1 = *(const v4f*)(p + 4);
  v4f f2 = *(const v4f*)(p + 16);
  v4f f3 = *(const v4f*)(p + 20);
  v16h b;
#pragma unroll
  for (int i = 0; i < 4; ++i) {
    b[i]      = (_Float16)f0[i];
    b[4 + i]  = (_Float16)f1[i];
    b[8 + i]  = (_Float16)f2[i];
    b[12 + i] = (_Float16)f3[i];
  }
  return b;
}

// B = W^T from pre-packed f16 weights [rowsP, ldw], always in-bounds
// (rows padded to mult 16, K padded to mult 32) -> 2x b128 loads
__device__ __forceinline__ v16h b_h(const _Float16* __restrict__ W, int ldw,
                                    int col0, int k0) {
  int lane = threadIdx.x & 31;
  const _Float16* p = W + (size_t)(col0 + (lane & 15)) * ldw + k0 + ((lane >> 4) << 3);
  v8h c0 = *(const v8h*)(p);
  v8h c1 = *(const v8h*)(p + 16);
  v16h b;
#pragma unroll
  for (int i = 0; i < 8; ++i) { b[i] = c0[i]; b[8 + i] = c1[i]; }
  return b;
}

// B row-major KxN f32 source (V matrix), fully in-bounds, strided loads
__device__ __forceinline__ v16h b_rm_nb(const float* __restrict__ Bm, int ldb,
                                        int k0, int col0) {
  int lane = threadIdx.x & 31;
  int n = col0 + (lane & 15);
  int kg = (lane >> 4) << 3;
  const float* p = Bm + (size_t)(k0 + kg) * ldb + n;
  v16h b;
#pragma unroll
  for (int i = 0; i < 16; ++i) b[i] = (_Float16)p[(size_t)((i & 7) + ((i >> 3) << 4)) * ldb];
  return b;
}

// A from LDS f16 tile (16 rows, ld mult of 32, 16B-aligned): 2x ds_load_b128
__device__ __forceinline__ v16h a_lds(const _Float16* __restrict__ S, int ld,
                                      int k0) {
  int lane = threadIdx.x & 31;
  const _Float16* p = S + (lane & 15) * ld + k0 + ((lane >> 4) << 3);
  v8h c0 = *(const v8h*)(p);
  v8h c1 = *(const v8h*)(p + 16);
  v16h a;
#pragma unroll
  for (int i = 0; i < 8; ++i) { a[i] = c0[i]; a[8 + i] = c1[i]; }
  return a;
}

// ---------------------------------------------------------------------------
// Elementwise / layout kernels
// ---------------------------------------------------------------------------
__global__ void k_zero(float* __restrict__ p, int n) {
  int t = blockIdx.x * blockDim.x + threadIdx.x;
  if (t < n) p[t] = 0.f;
}

// pack f32 W[rows,K] -> f16 [rowsP, Kp], zero padded
__global__ void k_pack_wh(const float* __restrict__ W, _Float16* __restrict__ o,
                          int rows, int K, int rowsP, int Kp, int tot) {
  int t = blockIdx.x * blockDim.x + threadIdx.x;
  if (t >= tot) return;
  int r = t / Kp, k = t - r * Kp;
  float v = (r < rows && k < K) ? W[(size_t)r * K + k] : 0.f;
  o[t] = (_Float16)v;
}

__global__ void k_bcn_to_bnc(const float* __restrict__ in, float* __restrict__ out,
                             int B, int C, int N, int tot) {
  int t = blockIdx.x * blockDim.x + threadIdx.x;
  if (t >= tot) return;
  int b = t / (N * C);
  int r = t - b * N * C;
  int n = r / C, c = r - n * C;
  out[t] = in[((size_t)b * C + c) * N + n];
}

__global__ void k_bnc_to_bcn(const float* __restrict__ in, float* __restrict__ out,
                             int B, int N, int C, int tot) {
  int t = blockIdx.x * blockDim.x + threadIdx.x;
  if (t >= tot) return;
  int b = t / (C * N);
  int r = t - b * C * N;
  int c = r / N, n = r - c * N;
  out[t] = in[((size_t)b * N + n) * C + c];
}

// ---------------------------------------------------------------------------
// Farthest point sampling: one block per batch, N<=4096, 256 threads
// ---------------------------------------------------------------------------
__global__ void k_fps(const float* __restrict__ xyzT, int N, int S,
                      int* __restrict__ out) {
  int b = blockIdx.x;
  const float* X = xyzT + (size_t)b * N * 3;
  __shared__ float sv[256];
  __shared__ int   si[256];
  __shared__ float sc[3];
  __shared__ int   sfar;
  int tid = threadIdx.x;
  float dist[16], px[16], py[16], pz[16];
  int cnt = 0;
  for (int j = tid; j < N; j += 256) {
    px[cnt] = X[j * 3 + 0]; py[cnt] = X[j * 3 + 1]; pz[cnt] = X[j * 3 + 2];
    dist[cnt] = 1e10f; cnt++;
  }
  if (tid == 0) sfar = 0;
  __syncthreads();
  for (int i = 0; i < S; ++i) {
    if (tid == 0) {
      int f = sfar;
      out[(size_t)b * S + i] = f;
      sc[0] = X[f * 3 + 0]; sc[1] = X[f * 3 + 1]; sc[2] = X[f * 3 + 2];
    }
    __syncthreads();
    float bx = sc[0], by = sc[1], bz = sc[2];
    float best = -1.f; int bi = 0; int c2 = 0;
    for (int j = tid; j < N; j += 256) {
      float dx = px[c2] - bx, dy = py[c2] - by, dz = pz[c2] - bz;
      float d = dx * dx + dy * dy + dz * dz;
      float nd = fminf(dist[c2], d);
      dist[c2] = nd;
      if (nd > best) { best = nd; bi = j; }
      c2++;
    }
    sv[tid] = best; si[tid] = bi;
    __syncthreads();
    for (int o = 128; o > 0; o >>= 1) {
      if (tid < o) {
        if (sv[tid + o] > sv[tid] ||
            (sv[tid + o] == sv[tid] && si[tid + o] < si[tid])) {
          sv[tid] = sv[tid + o]; si[tid] = si[tid + o];
        }
      }
      __syncthreads();
    }
    if (tid == 0) sfar = si[0];
    __syncthreads();
  }
}

__global__ void k_gather3(const float* __restrict__ xyzT, const int* __restrict__ idx,
                          float* __restrict__ out, int Np, int S, int tot) {
  int t = blockIdx.x * blockDim.x + threadIdx.x;
  if (t >= tot) return;
  int b = t / S;
  int j = idx[t];
  const float* src = xyzT + ((size_t)b * Np + j) * 3;
  out[(size_t)t * 3 + 0] = src[0];
  out[(size_t)t * 3 + 1] = src[1];
  out[(size_t)t * 3 + 2] = src[2];
}

__global__ void k_ballquery(const float* __restrict__ xyzT,
                            const float* __restrict__ cent,
                            int* __restrict__ gidx, int N, int S, float r2,
                            int tot) {
  int t = blockIdx.x * blockDim.x + threadIdx.x;
  if (t >= tot) return;
  int b = t / S;
  const float* X = xyzT + (size_t)b * N * 3;
  float cx = cent[(size_t)t * 3 + 0];
  float cy = cent[(size_t)t * 3 + 1];
  float cz = cent[(size_t)t * 3 + 2];
  int* o = gidx + (size_t)t * 64;
  int cnt = 0;
  for (int j = 0; j < N && cnt < 64; ++j) {
    float dx = X[j * 3 + 0] - cx, dy = X[j * 3 + 1] - cy, dz = X[j * 3 + 2] - cz;
    float d = dx * dx + dy * dy + dz * dz;
    if (d <= r2) o[cnt++] = j;
  }
  int pad = (cnt > 0) ? o[0] : 0;
  for (; cnt < 64; ++cnt) o[cnt] = pad;
}

// ---------------------------------------------------------------------------
// Fused set-abstraction: gather 16 samples -> 3x (1x1 conv + BN + ReLU) via
// LDS f16 tiles + WMMA (packed f16 weights) -> atomic max-pool.
// grid (4 rowtiles, S, B), block 256 (8 waves).
// ---------------------------------------------------------------------------
__device__ __forceinline__ void sa_layer(const _Float16* __restrict__ sIn, int ldin,
                                         const _Float16* __restrict__ Wp,
                                         const float* __restrict__ cb,
                                         const float* __restrict__ g,
                                         const float* __restrict__ bb, int Cout,
                                         _Float16* __restrict__ sOut) {
  int tid = threadIdx.x, wid = tid >> 5, lane = tid & 31, nw = blockDim.x >> 5;
  int hi = (lane >> 4) << 3;
  for (int tn = wid; tn < (Cout >> 4); tn += nw) {
    v8f acc = {};
    for (int k0 = 0; k0 < ldin; k0 += 32) {
      v16h a  = a_lds(sIn, ldin, k0);
      v16h bf = b_h(Wp, ldin, tn * 16, k0);
      acc = wmma32(a, bf, acc);
    }
    int n = tn * 16 + (lane & 15);
    float cbv = cb[n], gv = g[n] * BN_INV_STD, bv = bb[n];
#pragma unroll
    for (int r = 0; r < 8; ++r) {
      float v = fmaxf((acc[r] + cbv) * gv + bv, 0.f);
      sOut[(hi + r) * Cout + n] = (_Float16)v;
    }
  }
}

__global__ void k_sa_fused(const float* __restrict__ xyzT, int N,
                           const float* __restrict__ feats, int D,
                           const float* __restrict__ cent, int S,
                           const int* __restrict__ gidx,
                           const _Float16* w1, const float* cb1, const float* g1, const float* b1, int C1,
                           const _Float16* w2, const float* cb2, const float* g2, const float* b2, int C2,
                           const _Float16* w3, const float* cb3, const float* g3, const float* b3, int C3,
                           float* __restrict__ out) {
  __shared__ __align__(16) _Float16 sA[16 * 544];
  __shared__ __align__(16) _Float16 sB[16 * 512];
  int b = blockIdx.z, s = blockIdx.y, rt = blockIdx.x;
  int Cin = 3 + D;
  int CinP = (Cin + 31) & ~31;   // == packed K of layer-1 weights
  int tid = threadIdx.x;
  float cx = cent[((size_t)b * S + s) * 3 + 0];
  float cy = cent[((size_t)b * S + s) * 3 + 1];
  float cz = cent[((size_t)b * S + s) * 3 + 2];
  const int* gi = gidx + ((size_t)b * S + s) * 64 + rt * 16;
  for (int e = tid; e < 16 * CinP; e += blockDim.x) {
    int r = e / CinP, c = e - r * CinP;
    int j = gi[r];
    float v = 0.f;
    if (c < 3) {
      float p = xyzT[((size_t)b * N + j) * 3 + c];
      v = p - (c == 0 ? cx : (c == 1 ? cy : cz));
    } else if (c - 3 < D) {
      v = feats[((size_t)b * N + j) * D + (c - 3)];
    }
    sA[r * CinP + c] = (_Float16)v;
  }
  __syncthreads();
  sa_layer(sA, CinP, w1, cb1, g1, b1, C1, sB);
  __syncthreads();
  sa_layer(sB, C1, w2, cb2, g2, b2, C2, sA);
  __syncthreads();
  // layer 3 + partial max over 16 rows -> atomic max (post-ReLU >= 0)
  int wid = tid >> 5, lane = tid & 31, nw = blockDim.x >> 5;
  for (int tn = wid; tn < (C3 >> 4); tn += nw) {
    v8f acc = {};
    for (int k0 = 0; k0 < C2; k0 += 32) {
      v16h a  = a_lds(sA, C2, k0);
      v16h bf = b_h(w3, C2, tn * 16, k0);
      acc = wmma32(a, bf, acc);
    }
    int n = tn * 16 + (lane & 15);
    float cbv = cb3[n], gv = g3[n] * BN_INV_STD, bv = b3[n];
    float mx = 0.f;
#pragma unroll
    for (int r = 0; r < 8; ++r) {
      float v = fmaxf((acc[r] + cbv) * gv + bv, 0.f);
      mx = fmaxf(mx, v);
    }
    mx = fmaxf(mx, __shfl_xor(mx, 16, 32));
    if (lane < 16) {
      atomicMax(reinterpret_cast<unsigned int*>(
                    &out[((size_t)b * S + s) * C3 + n]),
                __float_as_uint(mx));
    }
  }
}

// ---------------------------------------------------------------------------
// Generic WMMA GEMM with pre-packed f16 weights:
// C[M,ldc] = act((A[M,K] * Wp[Np16,K]^T + bias) * g*INV_STD + b), 16x64 strip
// per wave. grid (ceil(N/64), M/16). K is a multiple of 32 at all call sites.
// ---------------------------------------------------------------------------
__global__ void k_gemm(const float* __restrict__ A, const _Float16* __restrict__ W,
                       const float* __restrict__ bias, const float* __restrict__ bn_g,
                       const float* __restrict__ bn_b, float* __restrict__ C,
                       int M, int K, int N, int ldc, int relu) {
  int row0 = blockIdx.y * 16;
  int colBase = blockIdx.x * 64;
  int nt = (N - colBase + 15) >> 4;
  if (nt > 4) nt = 4;
  bool fastA = (row0 + 16 <= M) && ((K & 3) == 0);
  v8f acc[4] = {{}, {}, {}, {}};
  for (int k0 = 0; k0 < K; k0 += 32) {
    v16h a = (fastA && (k0 + 32 <= K)) ? a_vec(A, K, row0, k0)
                                       : a_safe(A, K, row0, k0, M, K);
    if (k0 + 32 < K)
      __builtin_prefetch(A + (size_t)(row0 + (threadIdx.x & 15)) * K + k0 + 32, 0, 3);
#pragma unroll
    for (int t = 0; t < 4; ++t) {
      if (t < nt) {
        v16h bf = b_h(W, K, colBase + t * 16, k0);
        acc[t] = wmma32(a, bf, acc[t]);
      }
    }
  }
  int lane = threadIdx.x & 31;
  int hi = (lane >> 4) << 3;
#pragma unroll
  for (int t = 0; t < 4; ++t) {
    if (t < nt) {
      int n = colBase + t * 16 + (lane & 15);
      if (n < N) {
        float bs = bias ? bias[n] : 0.f;
        float gv = bn_g ? bn_g[n] * BN_INV_STD : 1.f;
        float bb = bn_b ? bn_b[n] : 0.f;
#pragma unroll
        for (int r = 0; r < 8; ++r) {
          int m = row0 + hi + r;
          if (m < M) {
            float v = (acc[t][r] + bs) * gv + bb;
            if (relu) v = fmaxf(v, 0.f);
            C[(size_t)m * ldc + n] = v;
          }
        }
      }
    }
  }
}

// ---------------------------------------------------------------------------
// Streaming attention. Q/K stored with padded leading dim dp (mult of 32,
// zero-filled pad columns), so every fragment load is the aligned vector path.
// ---------------------------------------------------------------------------
__global__ void k_attn_stats(const float* __restrict__ Q, const float* __restrict__ Km,
                             float* __restrict__ rmax, float* __restrict__ rsum,
                             int N, int dp) {
  int b = blockIdx.y, row0 = blockIdx.x * 16;
  const float* Qb = Q + (size_t)b * N * dp;
  const float* Kb = Km + (size_t)b * N * dp;
  int lane = threadIdx.x & 31;
  int hi = (lane >> 4) << 3;
  int nk = dp >> 5;
  v16h qf[4];
#pragma unroll
  for (int i = 0; i < 4; ++i)
    if (i < nk) qf[i] = a_vec(Qb, dp, row0, i * 32);
  float rm[8];
#pragma unroll
  for (int r = 0; r < 8; ++r) rm[r] = -3e38f;
  for (int kt = 0; kt < N; kt += 16) {
    v8f acc = {};
#pragma unroll
    for (int i = 0; i < 4; ++i)
      if (i < nk) acc = wmma32(qf[i], b_wT_vec(Kb, dp, kt, i * 32), acc);
#pragma unroll
    for (int r = 0; r < 8; ++r) rm[r] = fmaxf(rm[r], acc[r]);
  }
#pragma unroll
  for (int r = 0; r < 8; ++r)
    for (int o = 1; o < 16; o <<= 1) rm[r] = fmaxf(rm[r], __shfl_xor(rm[r], o, 32));
  float rs[8];
#pragma unroll
  for (int r = 0; r < 8; ++r) rs[r] = 0.f;
  for (int kt = 0; kt < N; kt += 16) {
    v8f acc = {};
#pragma unroll
    for (int i = 0; i < 4; ++i)
      if (i < nk) acc = wmma32(qf[i], b_wT_vec(Kb, dp, kt, i * 32), acc);
#pragma unroll
    for (int r = 0; r < 8; ++r) rs[r] += __expf(acc[r] - rm[r]);
  }
#pragma unroll
  for (int r = 0; r < 8; ++r)
    for (int o = 1; o < 16; o <<= 1) rs[r] += __shfl_xor(rs[r], o, 32);
  if ((lane & 15) == 0) {
#pragma unroll
    for (int r = 0; r < 8; ++r) {
      rmax[(size_t)b * N + row0 + hi + r] = rm[r];
      rsum[(size_t)b * N + row0 + hi + r] = rs[r];
    }
  }
}

// Y[b,n,c] = gamma * (softmax(QK^T) V)[n,c] + X[b,n,c]
// grid (C/16, N/16, B), block 32; P tile C-layout -> A-layout via LDS
__global__ void k_attn_out(const float* __restrict__ Q, const float* __restrict__ Km,
                           const float* __restrict__ V, const float* __restrict__ rmax,
                           const float* __restrict__ rsum, const float* __restrict__ gamma,
                           const float* __restrict__ X, float* __restrict__ Y,
                           int N, int dp, int C) {
  __shared__ __align__(16) _Float16 sP[16 * 32];
  int b = blockIdx.z, row0 = blockIdx.y * 16, col0 = blockIdx.x * 16;
  const float* Qb = Q + (size_t)b * N * dp;
  const float* Kb = Km + (size_t)b * N * dp;
  const float* Vb = V + (size_t)b * N * C;
  int lane = threadIdx.x & 31;
  int hi = (lane >> 4) << 3;
  int nk = dp >> 5;
  v16h qf[4];
#pragma unroll
  for (int i = 0; i < 4; ++i)
    if (i < nk) qf[i] = a_vec(Qb, dp, row0, i * 32);
  float rm[8], rs[8];
#pragma unroll
  for (int r = 0; r < 8; ++r) {
    rm[r] = rmax[(size_t)b * N + row0 + hi + r];
    rs[r] = rsum[(size_t)b * N + row0 + hi + r];
  }
  v8f acco = {};
  for (int kt = 0; kt < N; kt += 32) {
#pragma unroll
    for (int h = 0; h < 2; ++h) {
      int key0 = kt + h * 16;
      v8f acc = {};
#pragma unroll
      for (int i = 0; i < 4; ++i)
        if (i < nk) acc = wmma32(qf[i], b_wT_vec(Kb, dp, key0, i * 32), acc);
#pragma unroll
      for (int r = 0; r < 8; ++r) {
        float p = __expf(acc[r] - rm[r]) / rs[r];
        sP[(hi + r) * 32 + h * 16 + (lane & 15)] = (_Float16)p;
      }
    }
    __syncthreads();
    v16h a  = a_lds(sP, 32, 0);
    v16h bf = b_rm_nb(Vb, C, kt, col0);
    acco = wmma32(a, bf, acco);
    __syncthreads();
  }
  float gam = gamma[0];
  int n = col0 + (lane & 15);
#pragma unroll
  for (int r = 0; r < 8; ++r) {
    int m = row0 + hi + r;
    size_t o = ((size_t)b * N + m) * C + n;
    Y[o] = gam * acco[r] + X[o];
  }
}

// ---------------------------------------------------------------------------
// Feature propagation: 3-NN + inverse-distance weights, then interp+concat
// ---------------------------------------------------------------------------
__global__ void k_three_nn(const float* __restrict__ x1, const float* __restrict__ x2,
                           int* __restrict__ idx3, float* __restrict__ w3,
                           int N, int S, int tot) {
  int t = blockIdx.x * blockDim.x + threadIdx.x;
  if (t >= tot) return;
  int b = t / N;
  float ax = x1[(size_t)t * 3 + 0], ay = x1[(size_t)t * 3 + 1], az = x1[(size_t)t * 3 + 2];
  const float* Bp = x2 + (size_t)b * S * 3;
  float d0 = 3e38f, d1 = 3e38f, d2 = 3e38f;
  int i0 = 0, i1 = 0, i2 = 0;
  for (int j = 0; j < S; ++j) {
    float dx = ax - Bp[j * 3 + 0], dy = ay - Bp[j * 3 + 1], dz = az - Bp[j * 3 + 2];
    float d = dx * dx + dy * dy + dz * dz;
    if (d < d0)      { d2 = d1; i2 = i1; d1 = d0; i1 = i0; d0 = d; i0 = j; }
    else if (d < d1) { d2 = d1; i2 = i1; d1 = d; i1 = j; }
    else if (d < d2) { d2 = d; i2 = j; }
  }
  float w0 = 1.f / (d0 + 1e-8f), w1 = 1.f / (d1 + 1e-8f), w2 = 1.f / (d2 + 1e-8f);
  float ws = w0 + w1 + w2;
  idx3[(size_t)t * 3 + 0] = i0; idx3[(size_t)t * 3 + 1] = i1; idx3[(size_t)t * 3 + 2] = i2;
  w3[(size_t)t * 3 + 0] = w0 / ws; w3[(size_t)t * 3 + 1] = w1 / ws; w3[(size_t)t * 3 + 2] = w2 / ws;
}

__global__ void k_interp_concat(const float* __restrict__ p1, int C1,
                                const float* __restrict__ p2, int C2,
                                const int* __restrict__ idx3, const float* __restrict__ w3,
                                float* __restrict__ out, int N, int S, int Ct, int tot) {
  int t = blockIdx.x * blockDim.x + threadIdx.x;
  if (t >= tot) return;
  int c = t % Ct;
  int rem = t / Ct;
  int n = rem % N;
  int b = rem / N;
  float v;
  if (c < C1) {
    v = p1[((size_t)b * N + n) * C1 + c];
  } else {
    int c2 = c - C1;
    const int* id = idx3 + ((size_t)b * N + n) * 3;
    const float* w = w3 + ((size_t)b * N + n) * 3;
    v = w[0] * p2[((size_t)b * S + id[0]) * C2 + c2]
      + w[1] * p2[((size_t)b * S + id[1]) * C2 + c2]
      + w[2] * p2[((size_t)b * S + id[2]) * C2 + c2];
  }
  out[t] = v;
}

// ---------------------------------------------------------------------------
// GroupNorm (+ReLU) over [B,N,C] channels-last
// ---------------------------------------------------------------------------
__global__ void k_gn_relu(float* __restrict__ x, const float* __restrict__ gw,
                          const float* __restrict__ gb, int N, int C, int G) {
  int g = blockIdx.x, b = blockIdx.y;
  int Cg = C / G;
  size_t base = (size_t)b * N * C;
  int tid = threadIdx.x;
  int tot = N * Cg;
  float s = 0.f, ss = 0.f;
  for (int e = tid; e < tot; e += blockDim.x) {
    int n = e / Cg, c = g * Cg + (e - (e / Cg) * Cg);
    float v = x[base + (size_t)n * C + c];
    s += v; ss += v * v;
  }
  __shared__ float s1[256], s2[256];
  s1[tid] = s; s2[tid] = ss;
  __syncthreads();
  for (int o = 128; o > 0; o >>= 1) {
    if (tid < o) { s1[tid] += s1[tid + o]; s2[tid] += s2[tid + o]; }
    __syncthreads();
  }
  float mean = s1[0] / (float)tot;
  float var = s2[0] / (float)tot - mean * mean;
  float rstd = rsqrtf(var + 1e-5f);
  for (int e = tid; e < tot; e += blockDim.x) {
    int n = e / Cg, c = g * Cg + (e - (e / Cg) * Cg);
    size_t o = base + (size_t)n * C + c;
    float v = (x[o] - mean) * rstd * gw[c] + gb[c];
    x[o] = fmaxf(v, 0.f);
  }
}

// ---------------------------------------------------------------------------
// Host orchestration
// ---------------------------------------------------------------------------
extern "C" void kernel_launch(void* const* d_in, const int* in_sizes, int n_in,
                              void* d_out, int out_size, void* d_ws, size_t ws_size,
                              hipStream_t stream) {
  (void)in_sizes; (void)n_in; (void)out_size; (void)ws_size;
  const int B = 4, N0 = 4096;
  int cur = 0;
  auto nx = [&]() { return reinterpret_cast<const float*>(d_in[cur++]); };

  struct AttnP { const float *qw,*qb,*kw,*kb,*vw,*vb,*gamma; };
  struct MlpP  { const float *cw[3],*cb[3],*bg[3],*bb[3]; };
  auto getMlp = [&](int nl) {
    MlpP m{};
    for (int i = 0; i < nl; ++i) { m.cw[i] = nx(); m.cb[i] = nx(); }   // convs: w,b
    for (int i = 0; i < nl; ++i) { m.bg[i] = nx(); m.bb[i] = nx(); }   // bns: g,b
    return m;
  };
  auto getAttn = [&]() {
    AttnP a;
    a.qw = nx(); a.qb = nx(); a.kw = nx(); a.kb = nx();
    a.vw = nx(); a.vb = nx(); a.gamma = nx();
    return a;
  };

  // setup_inputs() insertion order: xyz, then params as written
  const float* xyz = nx();
  MlpP sa1 = getMlp(3); AttnP attn1 = getAttn();
  MlpP sa2 = getMlp(3); AttnP attn2 = getAttn();
  MlpP sa3 = getMlp(3); AttnP attn3 = getAttn();
  MlpP sa4 = getMlp(3); AttnP attn4 = getAttn();
  MlpP fp4 = getMlp(2); AttnP fp4a = getAttn();
  MlpP fp3 = getMlp(2); AttnP fp3a = getAttn();
  MlpP fp2 = getMlp(2); AttnP fp2a = getAttn();
  MlpP fp1 = getMlp(3); AttnP fp1a = getAttn();
  const float* c1w = nx(); const float* c1b = nx();
  const float* gn1g = nx(); const float* gn1b = nx();
  const float* c2w = nx(); const float* c2b = nx();
  const float* gn2g = nx(); const float* gn2b = nx();
  const float* c3w = nx(); const float* c3b = nx();
  const float* gn3g = nx(); const float* gn3b = nx();
  const float* c4w = nx(); const float* c4b = nx();

  size_t off = 0;
  auto alloc = [&](size_t n) {
    float* p = (float*)d_ws + off;
    off += (n + 63) & ~(size_t)63;
    return p;
  };

  // ---- pack every conv weight once into zero-padded f16 [rowsP, Kp] ----
  struct PW { const _Float16* p; int ld; };
  auto pack = [&](const float* w, int rows, int K) {
    int Kp = (K + 31) & ~31;
    int rowsP = (rows + 15) & ~15;
    _Float16* o = (_Float16*)alloc(((size_t)rowsP * Kp + 1) / 2);
    int tot = rowsP * Kp;
    k_pack_wh<<<(tot + 255) / 256, 256, 0, stream>>>(w, o, rows, K, rowsP, Kp, tot);
    return PW{o, Kp};
  };
  struct MlpK  { PW w[3]; const float *cb[3],*bg[3],*bb[3]; };
  auto packMlp = [&](const MlpP& m, int nl, int cin, const int* chans) {
    MlpK k{};
    int last = cin;
    for (int i = 0; i < nl; ++i) {
      k.w[i] = pack(m.cw[i], chans[i], last);
      k.cb[i] = m.cb[i]; k.bg[i] = m.bg[i]; k.bb[i] = m.bb[i];
      last = chans[i];
    }
    return k;
  };
  struct AttnK { PW qw,kw,vw; const float *qb,*kb,*vb,*gamma; };
  auto packAttn = [&](const AttnP& a, int C) {
    AttnK t;
    t.qw = pack(a.qw, C / 8, C); t.kw = pack(a.kw, C / 8, C);
    t.vw = pack(a.vw, C, C);
    t.qb = a.qb; t.kb = a.kb; t.vb = a.vb; t.gamma = a.gamma;
    return t;
  };

  int ch_sa1[3] = {64, 64, 128},  ch_sa2[3] = {128, 128, 256};
  int ch_sa3[3] = {256, 256, 512}, ch_sa4[3] = {512, 512, 1024};
  int ch_fp4[2] = {512, 512}, ch_fp3[2] = {512, 512};
  int ch_fp2[2] = {512, 256}, ch_fp1[3] = {256, 128, 128};
  MlpK sa1k = packMlp(sa1, 3, 3, ch_sa1);
  MlpK sa2k = packMlp(sa2, 3, 131, ch_sa2);
  MlpK sa3k = packMlp(sa3, 3, 259, ch_sa3);
  MlpK sa4k = packMlp(sa4, 3, 515, ch_sa4);
  MlpK fp4k = packMlp(fp4, 2, 1536, ch_fp4);
  MlpK fp3k = packMlp(fp3, 2, 768, ch_fp3);
  MlpK fp2k = packMlp(fp2, 2, 640, ch_fp2);
  MlpK fp1k = packMlp(fp1, 3, 256, ch_fp1);
  AttnK a1k = packAttn(attn1, 128), a2k = packAttn(attn2, 256);
  AttnK a3k = packAttn(attn3, 512), a4k = packAttn(attn4, 1024);
  AttnK f4ak = packAttn(fp4a, 512), f3ak = packAttn(fp3a, 512);
  AttnK f2ak = packAttn(fp2a, 256), f1ak = packAttn(fp1a, 128);
  PW pc1 = pack(c1w, 256, 128), pc2 = pack(c2w, 128, 256);
  PW pc3 = pack(c3w, 64, 128),  pc4 = pack(c4w, 13, 64);

  // persistent buffers
  float* xyzT = alloc((size_t)B * N0 * 3);
  float* xyz1 = alloc((size_t)B * 1024 * 3);
  float* xyz2 = alloc((size_t)B * 512 * 3);
  float* xyz3 = alloc((size_t)B * 128 * 3);
  float* xyz4 = alloc((size_t)B * 32 * 3);
  float* l1  = alloc((size_t)B * 1024 * 128);
  float* l2  = alloc((size_t)B * 512 * 256);
  float* l3  = alloc((size_t)B * 128 * 512);
  float* l4  = alloc((size_t)B * 32 * 1024);
  float* l3f = alloc((size_t)B * 128 * 512);
  float* l2f = alloc((size_t)B * 512 * 512);
  float* l1f = alloc((size_t)B * 1024 * 256);
  float* l0f = alloc((size_t)B * 4096 * 128);
  size_t scratch0 = off;

  auto gemm = [&](const float* A, PW W, const float* bias, const float* g,
                  const float* bb, float* C, int M, int K, int Nc, int ldc,
                  int relu) {
    dim3 gr((Nc + 63) / 64, (M + 15) / 16, 1);
    k_gemm<<<gr, 32, 0, stream>>>(A, W.p, bias, g, bb, C, M, K, Nc, ldc, relu);
  };

  auto attn = [&](const float* X, float* Y, int Np, int C, const AttnK& a) {
    int d = C / 8;
    int dp = (d < 32) ? 32 : d;
    float* q = alloc((size_t)B * Np * dp);
    float* k = alloc((size_t)B * Np * dp);
    float* v = alloc((size_t)B * Np * C);
    float* rmx = alloc((size_t)B * Np);
    float* rsm = alloc((size_t)B * Np);
    if (dp != d) {
      int zn = B * Np * dp;
      k_zero<<<(zn + 255) / 256, 256, 0, stream>>>(q, zn);
      k_zero<<<(zn + 255) / 256, 256, 0, stream>>>(k, zn);
    }
    gemm(X, a.qw, a.qb, nullptr, nullptr, q, B * Np, C, d, dp, 0);
    gemm(X, a.kw, a.kb, nullptr, nullptr, k, B * Np, C, d, dp, 0);
    gemm(X, a.vw, a.vb, nullptr, nullptr, v, B * Np, C, C, C, 0);
    k_attn_stats<<<dim3(Np / 16, B), 32, 0, stream>>>(q, k, rmx, rsm, Np, dp);
    k_attn_out<<<dim3(C / 16, Np / 16, B), 32, 0, stream>>>(q, k, v, rmx, rsm,
                                                            a.gamma, X, Y, Np, dp, C);
  };

  auto saStage = [&](const float* xyzPrev, int Np, const float* feat, int D,
                     float* xyzNew, int S, float radius, const MlpK& m,
                     int C1, int C2, int C3, float* pre) {
    int* fidx = (int*)alloc((size_t)B * S);
    int* gix  = (int*)alloc((size_t)B * S * 64);
    k_fps<<<B, 256, 0, stream>>>(xyzPrev, Np, S, fidx);
    int tot = B * S;
    k_gather3<<<(tot + 255) / 256, 256, 0, stream>>>(xyzPrev, fidx, xyzNew, Np, S, tot);
    k_ballquery<<<(tot + 255) / 256, 256, 0, stream>>>(xyzPrev, xyzNew, gix, Np, S,
                                                       radius * radius, tot);
    int zn = B * S * C3;
    k_zero<<<(zn + 255) / 256, 256, 0, stream>>>(pre, zn);
    dim3 gr(4, S, B);
    k_sa_fused<<<gr, 256, 0, stream>>>(xyzPrev, Np, feat, D, xyzNew, S, gix,
        m.w[0].p, m.cb[0], m.bg[0], m.bb[0], C1,
        m.w[1].p, m.cb[1], m.bg[1], m.bb[1], C2,
        m.w[2].p, m.cb[2], m.bg[2], m.bb[2], C3, pre);
  };

  auto fpStage = [&](const float* xyzA, int Na, const float* xyzB, int Sb,
                     const float* p1, int C1, const float* p2, int C2,
                     const MlpK& m, int nl, const int* chans, float* outPre) {
    int* idx3 = (int*)alloc((size_t)B * Na * 3);
    float* w3 = alloc((size_t)B * Na * 3);
    int tot = B * Na;
    k_three_nn<<<(tot + 255) / 256, 256, 0, stream>>>(xyzA, xyzB, idx3, w3, Na, Sb, tot);
    int Ct = C1 + C2;
    float* cc = alloc((size_t)B * Na * Ct);
    int tote = B * Na * Ct;
    k_interp_concat<<<(tote + 255) / 256, 256, 0, stream>>>(p1, C1, p2, C2, idx3, w3,
                                                            cc, Na, Sb, Ct, tote);
    const float* in = cc;
    int K = Ct;
    for (int i = 0; i < nl; ++i) {
      float* o = (i == nl - 1) ? outPre : alloc((size_t)B * Na * chans[i]);
      gemm(in, m.w[i], m.cb[i], m.bg[i], m.bb[i], o, B * Na, K, chans[i], chans[i], 1);
      in = o;
      K = chans[i];
    }
  };

  // ---------------- pipeline ----------------
  {
    int tot = B * N0 * 3;
    k_bcn_to_bnc<<<(tot + 255) / 256, 256, 0, stream>>>(xyz, xyzT, B, 3, N0, tot);
  }
  off = scratch0;
  { float* pre = alloc((size_t)B * 1024 * 128);
    saStage(xyzT, N0, nullptr, 0, xyz1, 1024, 0.1f, sa1k, 64, 64, 128, pre);
    attn(pre, l1, 1024, 128, a1k); }
  off = scratch0;
  { float* pre = alloc((size_t)B * 512 * 256);
    saStage(xyz1, 1024, l1, 128, xyz2, 512, 0.2f, sa2k, 128, 128, 256, pre);
    attn(pre, l2, 512, 256, a2k); }
  off = scratch0;
  { float* pre = alloc((size_t)B * 128 * 512);
    saStage(xyz2, 512, l2, 256, xyz3, 128, 0.4f, sa3k, 256, 256, 512, pre);
    attn(pre, l3, 128, 512, a3k); }
  off = scratch0;
  { float* pre = alloc((size_t)B * 32 * 1024);
    saStage(xyz3, 128, l3, 512, xyz4, 32, 0.8f, sa4k, 512, 512, 1024, pre);
    attn(pre, l4, 32, 1024, a4k); }

  off = scratch0;
  { float* p = alloc((size_t)B * 128 * 512);
    fpStage(xyz3, 128, xyz4, 32, l3, 512, l4, 1024, fp4k, 2, ch_fp4, p);
    attn(p, l3f, 128, 512, f4ak); }
  off = scratch0;
  { float* p = alloc((size_t)B * 512 * 512);
    fpStage(xyz2, 512, xyz3, 128, l2, 256, l3f, 512, fp3k, 2, ch_fp3, p);
    attn(p, l2f, 512, 512, f3ak); }
  off = scratch0;
  { float* p = alloc((size_t)B * 1024 * 256);
    fpStage(xyz1, 1024, xyz2, 512, l1, 128, l2f, 512, fp2k, 2, ch_fp2, p);
    attn(p, l1f, 1024, 256, f2ak); }
  off = scratch0;
  { float* p = alloc((size_t)B * 4096 * 128);
    fpStage(xyzT, 4096, xyz1, 1024, nullptr, 0, l1f, 256, fp1k, 3, ch_fp1, p);
    attn(p, l0f, 4096, 128, f1ak); }

  // head
  off = scratch0;
  float* y1 = alloc((size_t)B * 4096 * 256);
  gemm(l0f, pc1, c1b, nullptr, nullptr, y1, B * 4096, 128, 256, 256, 0);
  k_gn_relu<<<dim3(16, B), 256, 0, stream>>>(y1, gn1g, gn1b, 4096, 256, 16);
  float* y2 = alloc((size_t)B * 4096 * 128);
  gemm(y1, pc2, c2b, nullptr, nullptr, y2, B * 4096, 256, 128, 128, 0);
  k_gn_relu<<<dim3(8, B), 256, 0, stream>>>(y2, gn2g, gn2b, 4096, 128, 8);
  float* y3 = alloc((size_t)B * 4096 * 64);
  gemm(y2, pc3, c3b, nullptr, nullptr, y3, B * 4096, 128, 64, 64, 0);
  k_gn_relu<<<dim3(8, B), 256, 0, stream>>>(y3, gn3g, gn3b, 4096, 64, 8);
  float* y4 = alloc((size_t)B * 4096 * 13);
  gemm(y3, pc4, c4b, nullptr, nullptr, y4, B * 4096, 64, 13, 13, 0);
  {
    int tot = B * 13 * 4096;
    k_bnc_to_bcn<<<(tot + 255) / 256, 256, 0, stream>>>(y4, (float*)d_out, B, 4096, 13, tot);
  }
}
